// MultiHeadAttention_81509889343490
// MI455X (gfx1250) — matrix operand order repaired and checked by reading, not compiled, not run on hardware
//
#include <hip/hip_runtime.h>
#include <hip/hip_bf16.h>

// Problem constants (match reference)
#define BB  4
#define TT  2048
#define DD  1024
#define HH  16
#define DHH 64

typedef __attribute__((ext_vector_type(16))) __bf16 v16bf;
typedef __attribute__((ext_vector_type(8)))  __bf16 v8bf;
typedef __attribute__((ext_vector_type(8)))  float  v8f;

// Native fptrunc: backend selects v_cvt_pk_bf16_f32 on gfx1250.
static __device__ __forceinline__ __bf16 f2bf(float f) { return (__bf16)f; }

static __device__ __forceinline__ v16bf concat8(v8bf a, v8bf b) {
  return __builtin_shufflevector(a, b, 0,1,2,3,4,5,6,7,8,9,10,11,12,13,14,15);
}

static __device__ __forceinline__ v8f wmma_bf16(v16bf a, v16bf b, v8f c) {
  // D = A(16x32 bf16) * B(32x16 bf16) + C(16x16 f32)
  return __builtin_amdgcn_wmma_f32_16x16x32_bf16(false, a, false, b, (short)0, c,
                                                 false, false);
}

// ---------------------------------------------------------------------------
// Kernel 0: one-time fp32 -> bf16 conversion (x and the four weight matrices).
// Each thread converts 8 contiguous elements (32-byte fp32 read, one 16-byte bf16 store).
// ---------------------------------------------------------------------------
__global__ __launch_bounds__(256)
void mha_cvt_bf16_kernel(const float* __restrict__ src,
                         __bf16* __restrict__ dst, int n8) {
  const int i = blockIdx.x * blockDim.x + threadIdx.x;
  if (i >= n8) return;
  const float* s = src + (size_t)i * 8;
  v8bf r;
#pragma unroll
  for (int j = 0; j < 8; ++j) r[j] = f2bf(s[j]);
  *(v8bf*)(dst + (size_t)i * 8) = r;
}

// ---------------------------------------------------------------------------
// Kernel 1: fused QKV projection. One wave computes a 64x64 tile of X*W^T
// (4 token fragments x 4 feature fragments = 16 WMMAs per K-chunk).
// grid = (B*T/64, H, 3), block = 32 (one wave).
//   which=0 -> Q [B,H,T,DH] bf16
//   which=1 -> K [B,H,T,DH] bf16
//   which=2 -> V^T [B,H,DH,T] bf16   (transposed so PV B-operand is contiguous)
// ---------------------------------------------------------------------------
__global__ __launch_bounds__(32)
void mha_qkv_proj_kernel(const __bf16* __restrict__ xb,
                         const __bf16* __restrict__ Wqb,
                         const __bf16* __restrict__ Wkb,
                         const __bf16* __restrict__ Wvb,
                         __bf16* __restrict__ Qd,
                         __bf16* __restrict__ Kd,
                         __bf16* __restrict__ Vtd) {
  const int mt    = blockIdx.x;        // 64-token tile
  const int head  = blockIdx.y;        // 64 output cols = one head
  const int which = blockIdx.z;        // 0=Q 1=K 2=V
  const int lane  = threadIdx.x;
  const int ln    = lane & 15;
  const int hf    = lane >> 4;

  const __bf16* __restrict__ W = (which == 0) ? Wqb : (which == 1) ? Wkb : Wvb;
  const __bf16* __restrict__ xrow = xb + (size_t)(mt * 64 + ln) * DD;
  const __bf16* __restrict__ wrow = W + (size_t)(head * 64 + ln) * DD + 16 * hf;

  v8f acc[4][4] = {};                  // 64x64 f32 accumulator

  for (int kb = 0; kb < DD; kb += 32) {
    // A operands: token groups r*16; per-lane contiguous 16 at K = kb + 16h
    v16bf am[4];
#pragma unroll
    for (int r = 0; r < 4; ++r) {
      const __bf16* p = xrow + (size_t)(r * 16) * DD + kb;
      am[r] = concat8(*(const v8bf*)(p + 16 * hf),
                      *(const v8bf*)(p + 16 * hf + 8));
    }
#pragma unroll
    for (int f = 0; f < 4; ++f) {
      // B operand: column n = weight row; contiguous 16 at K = kb + 16h
      v16bf bw = *(const v16bf*)(wrow + (size_t)(f * 16) * DD + kb);
#pragma unroll
      for (int r = 0; r < 4; ++r)
        acc[r][f] = wmma_bf16(am[r], bw, acc[r][f]);
    }
  }

  // Store: C layout: lane -> col n (=f*16+ln), VGPR slot i -> row 8h+i
#pragma unroll
  for (int r = 0; r < 4; ++r) {
#pragma unroll
    for (int f = 0; f < 4; ++f) {
#pragma unroll
      for (int i = 0; i < 8; ++i) {
        const int m  = mt * 64 + r * 16 + 8 * hf + i;  // global token row
        const int b  = m >> 11;                        // / T
        const int t  = m & (TT - 1);
        const int dh = f * 16 + ln;
        const __bf16 v = f2bf(acc[r][f][i]);
        if (which == 0)
          Qd[(((size_t)b * HH + head) * TT + t) * DHH + dh] = v;
        else if (which == 1)
          Kd[(((size_t)b * HH + head) * TT + t) * DHH + dh] = v;
        else
          Vtd[(((size_t)b * HH + head) * DHH + dh) * (size_t)TT + t] = v;
      }
    }
  }
}

// ---------------------------------------------------------------------------
// Kernel 2: causal flash attention. One wave per 16-query tile per (b,h).
// grid = (T/16, H, B), block = 32.
// ---------------------------------------------------------------------------
__global__ __launch_bounds__(32)
void mha_flash_attn_kernel(const __bf16* __restrict__ Qd,
                           const __bf16* __restrict__ Kd,
                           const __bf16* __restrict__ Vtd,
                           __bf16* __restrict__ attn) {
  const int qt   = blockIdx.x;
  const int head = blockIdx.y;
  const int b    = blockIdx.z;
  const int lane = threadIdx.x;
  const int ln   = lane & 15;
  const int hf   = lane >> 4;
  const size_t bh = (size_t)b * HH + head;
  const int qbase = qt * 16;

  __shared__ alignas(16) __bf16 Pl[16 * 32];   // probability tile restage

  // Q A-operand (16 queries x DH=64 -> two K=32 chunks), loaded once
  const __bf16* qrow = Qd + (bh * TT + qbase + ln) * DHH;
  v16bf qa[2];
#pragma unroll
  for (int c = 0; c < 2; ++c)
    qa[c] = concat8(*(const v8bf*)(qrow + 32 * c + 16 * hf),
                    *(const v8bf*)(qrow + 32 * c + 16 * hf + 8));

  float m_i[8], l_i[8];
#pragma unroll
  for (int i = 0; i < 8; ++i) { m_i[i] = -3.0e38f; l_i[i] = 0.0f; }
  v8f o[4] = {};                                // 16 x 64 output accumulator

  const int nkb = (qbase + 15) / 32 + 1;        // causal bound, 32-key blocks
  for (int kb = 0; kb < nkb; ++kb) {
    const int kbase = kb * 32;

    // ---- S = Q * K^T : 16x32 scores = 2 N-fragments x 2 K-chunks ----
    v8f s[2];
#pragma unroll
    for (int g = 0; g < 2; ++g) {
      v8f a = {};
      const __bf16* krow = Kd + (bh * TT + (kbase + g * 16 + ln)) * DHH;
#pragma unroll
      for (int c = 0; c < 2; ++c)
        a = wmma_bf16(qa[c], *(const v16bf*)(krow + 32 * c + 16 * hf), a);
      s[g] = a;
    }

    // ---- online softmax (rows in VGPR slots, columns across lanes) ----
#pragma unroll
    for (int i = 0; i < 8; ++i) {
      const int q  = qbase + 8 * hf + i;
      const int k0 = kbase + ln;
      const int k1 = kbase + 16 + ln;
      float s0 = (k0 <= q) ? s[0][i] * 0.125f : -1.0e9f;   // 1/sqrt(64)
      float s1 = (k1 <= q) ? s[1][i] * 0.125f : -1.0e9f;

      float mx = fmaxf(s0, s1);
      mx = fmaxf(mx, __shfl_xor(mx, 1, 32));
      mx = fmaxf(mx, __shfl_xor(mx, 2, 32));
      mx = fmaxf(mx, __shfl_xor(mx, 4, 32));
      mx = fmaxf(mx, __shfl_xor(mx, 8, 32));

      const float mnew  = fmaxf(m_i[i], mx);
      const float alpha = __expf(m_i[i] - mnew);
      const float p0 = __expf(s0 - mnew);
      const float p1 = __expf(s1 - mnew);

      float rs = p0 + p1;
      rs += __shfl_xor(rs, 1, 32);
      rs += __shfl_xor(rs, 2, 32);
      rs += __shfl_xor(rs, 4, 32);
      rs += __shfl_xor(rs, 8, 32);

      l_i[i] = l_i[i] * alpha + rs;
      m_i[i] = mnew;
#pragma unroll
      for (int f = 0; f < 4; ++f) o[f][i] *= alpha;

      const int row = 8 * hf + i;
      Pl[row * 32 + ln]      = f2bf(p0);
      Pl[row * 32 + 16 + ln] = f2bf(p1);
    }
    __syncthreads();

    // ---- O += P * V : restage P as A-operand, V^T rows contiguous B ----
    v16bf pa = concat8(*(const v8bf*)(Pl + ln * 32 + 16 * hf),
                       *(const v8bf*)(Pl + ln * 32 + 16 * hf + 8));
#pragma unroll
    for (int f = 0; f < 4; ++f) {
      const __bf16* vrow =
          Vtd + (bh * DHH + f * 16 + ln) * (size_t)TT + kbase + 16 * hf;
      o[f] = wmma_bf16(pa, *(const v16bf*)vrow, o[f]);
    }
    __syncthreads();
  }

  // ---- normalize and emit bf16 attn output [B,T,D] (token-major) ----
#pragma unroll
  for (int i = 0; i < 8; ++i) {
    const int q = qbase + 8 * hf + i;
    const float inv = 1.0f / l_i[i];
#pragma unroll
    for (int f = 0; f < 4; ++f)
      attn[((size_t)b * TT + q) * DD + head * 64 + f * 16 + ln] =
          f2bf(o[f][i] * inv);
  }
}

// ---------------------------------------------------------------------------
// Kernel 3: output projection, fp32 result.  out = attn_bf16 * Wo_bf16^T
// One wave computes a 64x64 tile.  grid = (B*T/64, D/64), block = 32.
// ---------------------------------------------------------------------------
__global__ __launch_bounds__(32)
void mha_out_proj_kernel(const __bf16* __restrict__ attn,
                         const __bf16* __restrict__ Wob,
                         float* __restrict__ out) {
  const int mt   = blockIdx.x;
  const int nt   = blockIdx.y;
  const int lane = threadIdx.x;
  const int ln   = lane & 15;
  const int hf   = lane >> 4;

  const __bf16* __restrict__ arow = attn + (size_t)(mt * 64 + ln) * DD;
  const __bf16* __restrict__ wrow = Wob + (size_t)(nt * 64 + ln) * DD + 16 * hf;

  v8f acc[4][4] = {};
  for (int kb = 0; kb < DD; kb += 32) {
    v16bf am[4];
#pragma unroll
    for (int r = 0; r < 4; ++r) {
      const __bf16* p = arow + (size_t)(r * 16) * DD + kb;
      am[r] = concat8(*(const v8bf*)(p + 16 * hf),
                      *(const v8bf*)(p + 16 * hf + 8));
    }
#pragma unroll
    for (int f = 0; f < 4; ++f) {
      v16bf bw = *(const v16bf*)(wrow + (size_t)(f * 16) * DD + kb);
#pragma unroll
      for (int r = 0; r < 4; ++r)
        acc[r][f] = wmma_bf16(am[r], bw, acc[r][f]);
    }
  }

#pragma unroll
  for (int r = 0; r < 4; ++r) {
#pragma unroll
    for (int f = 0; f < 4; ++f) {
#pragma unroll
      for (int i = 0; i < 8; ++i) {
        const int m = mt * 64 + r * 16 + 8 * hf + i;
        const int n = nt * 64 + f * 16 + ln;
        out[(size_t)m * DD + n] = acc[r][f][i];
      }
    }
  }
}

// ---------------------------------------------------------------------------
extern "C" void kernel_launch(void* const* d_in, const int* in_sizes, int n_in,
                              void* d_out, int out_size, void* d_ws, size_t ws_size,
                              hipStream_t stream) {
  // inputs: x, mask, Wq, Wk, Wv, Wo  (mask is implicit-causal; unused)
  const float* x  = (const float*)d_in[0];
  const float* Wq = (const float*)d_in[2];
  const float* Wk = (const float*)d_in[3];
  const float* Wv = (const float*)d_in[4];
  const float* Wo = (const float*)d_in[5];
  float* out = (float*)d_out;

  const size_t btd = (size_t)BB * TT * DD;              // 8388608
  const size_t dd  = (size_t)DD * DD;                   // 1048576

  __bf16* Qd   = (__bf16*)d_ws;
  __bf16* Kd   = Qd + btd;
  __bf16* Vtd  = Kd + btd;
  __bf16* attn = Vtd + btd;
  __bf16* xb   = attn + btd;
  __bf16* Wqb  = xb + btd;
  __bf16* Wkb  = Wqb + dd;
  __bf16* Wvb  = Wkb + dd;
  __bf16* Wob  = Wvb + dd;                              // total ~92 MB

  // --- one-time fp32 -> bf16 conversions ---
  {
    const int n8x = (int)(btd / 8);
    mha_cvt_bf16_kernel<<<(n8x + 255) / 256, 256, 0, stream>>>(x, xb, n8x);
    const int n8w = (int)(dd / 8);
    const int blk = (n8w + 255) / 256;
    mha_cvt_bf16_kernel<<<blk, 256, 0, stream>>>(Wq, Wqb, n8w);
    mha_cvt_bf16_kernel<<<blk, 256, 0, stream>>>(Wk, Wkb, n8w);
    mha_cvt_bf16_kernel<<<blk, 256, 0, stream>>>(Wv, Wvb, n8w);
    mha_cvt_bf16_kernel<<<blk, 256, 0, stream>>>(Wo, Wob, n8w);
  }
  {
    dim3 grid(BB * TT / 64, HH, 3);
    mha_qkv_proj_kernel<<<grid, 32, 0, stream>>>(xb, Wqb, Wkb, Wvb, Qd, Kd, Vtd);
  }
  {
    dim3 grid(TT / 16, HH, BB);
    mha_flash_attn_kernel<<<grid, 32, 0, stream>>>(Qd, Kd, Vtd, attn);
  }
  {
    dim3 grid(BB * TT / 64, DD / 64);
    mha_out_proj_kernel<<<grid, 32, 0, stream>>>(attn, Wob, out);
  }
}
